// edge_predictor_66116726555434
// MI455X (gfx1250) — matrix operand order repaired and checked by reading, not compiled, not run on hardware
//
#include <hip/hip_runtime.h>
#include <hip/hip_bf16.h>

typedef __attribute__((ext_vector_type(16))) _Float16 v16h;
typedef __attribute__((ext_vector_type(8)))  float    v8f;

// ---------------- elementwise / graph kernels ----------------

__global__ void fill_kernel(float* __restrict__ p, float v, long n) {
  long t = (long)blockIdx.x * blockDim.x + threadIdx.x;
  if (t < n) p[t] = v;
}

__global__ void deg_kernel(const int* __restrict__ dst, float* __restrict__ deg, int E) {
  int t = blockIdx.x * blockDim.x + threadIdx.x;
  if (t < E) atomicAdd(&deg[dst[t]], 1.0f);
}

__global__ void rsqrt_kernel(float* __restrict__ p, int n) {
  int t = blockIdx.x * blockDim.x + threadIdx.x;
  if (t < n) p[t] = rsqrtf(p[t]);
}

// C[M,Nc] = A[M,K] @ W[K,Nc] (+ bias), W staged in LDS
__global__ void dense_kernel(const float* __restrict__ A, const float* __restrict__ W,
                             const float* __restrict__ bias, float* __restrict__ C,
                             int M, int K, int Nc, int addBias) {
  extern __shared__ float sw[];
  for (int i = threadIdx.x; i < K * Nc; i += blockDim.x) sw[i] = W[i];
  __syncthreads();
  long t = (long)blockIdx.x * blockDim.x + threadIdx.x;
  if (t >= (long)M * Nc) return;
  int i = (int)(t / Nc), j = (int)(t % Nc);
  const float* ar = A + (long)i * K;
  float s = addBias ? bias[j] : 0.0f;
  for (int k = 0; k < K; ++k) s += ar[k] * sw[k * Nc + j];
  C[t] = s;
}

// one wave32 per edge: acc[dst] += dinv[src]*dinv[dst] * xw[src]
__global__ void scatter_kernel(const int* __restrict__ src, const int* __restrict__ dst,
                               const float* __restrict__ dinv, const float* __restrict__ xw,
                               float* __restrict__ acc, int F, int E) {
  int e = (blockIdx.x * blockDim.x + threadIdx.x) >> 5;
  int lane = threadIdx.x & 31;
  if (e >= E) return;
  int s = src[e], d = dst[e];
  float w = dinv[s] * dinv[d];
  const float* xr = xw + (long)s * F;
  float* ac = acc + (long)d * F;
  for (int k = lane; k < F; k += 32) atomicAdd(&ac[k], w * xr[k]);
}

// fold self-loop contribution + bias (no atomics needed)
__global__ void finish_kernel(float* __restrict__ acc, const float* __restrict__ xw,
                              const float* __restrict__ dinv, const float* __restrict__ bias,
                              int M, int F) {
  long t = (long)blockIdx.x * blockDim.x + threadIdx.x;
  if (t >= (long)M * F) return;
  int i = (int)(t / F), k = (int)(t % F);
  float di = dinv[i];
  acc[t] += xw[t] * di * di + bias[k];
}

// ---------------- fused gather + 3-layer MLP (WMMA f16->f32) ----------------
// Padded shapes: [16x128]@[128x48] -> [16x64]@[64x32] -> [16x32]@[32x16]
// A 16x32 f16 frag: lane half h, row m=lane%16, a[j]=K(kb + h*8 + (j<8? j : j+8))
// B 32x16 f16 frag: lane col n=lane%16, b[j]=K(kb + h*8 + (j<8? j : j+8))
// C/D 16x16 f32:    c[v] = D[row v + 8*h][col n]
// Weights are pre-swizzled in LDS into fragment-major layout [tile][lane][16]
// so every B fragment is 32 contiguous bytes -> 2x ds_load_b128.
__global__ __launch_bounds__(256)
void mlp_wmma_kernel(const float* __restrict__ allh,
                     const int* __restrict__ node1, const int* __restrict__ node2,
                     const float* __restrict__ Wf1, const float* __restrict__ bf1,
                     const float* __restrict__ Wf2, const float* __restrict__ bf2,
                     const float* __restrict__ Wf3, const float* __restrict__ bf3,
                     float* __restrict__ zout, int P) {
  __shared__ _Float16 w1s[12 * 512];   // 12 tiles (4 kt x 3 nt), frag-major
  __shared__ _Float16 w2s[4 * 512];    // 4 tiles (2 kt x 2 nt)
  __shared__ _Float16 w3s[1 * 512];    // 1 tile
  __shared__ float    bias1[48];
  __shared__ float    bias2[32];
  __shared__ _Float16 ztile[8][16 * 128];

  const int tid = threadIdx.x;
  // ---- stage weights, swizzled to B-fragment layout ----
  for (int idx = tid; idx < 12 * 512; idx += 256) {
    int tile = idx >> 9, rem = idx & 511;
    int ln = rem >> 4, j = rem & 15;
    int kt = tile / 3, nt = tile % 3;
    int k = kt * 32 + (ln >> 4) * 8 + (j < 8 ? j : j + 8);
    int n = nt * 16 + (ln & 15);
    w1s[idx] = (_Float16)((k < 100 && n < 40) ? Wf1[k * 40 + n] : 0.0f);
  }
  for (int idx = tid; idx < 4 * 512; idx += 256) {
    int tile = idx >> 9, rem = idx & 511;
    int ln = rem >> 4, j = rem & 15;
    int kt = tile >> 1, nt = tile & 1;
    int k = kt * 32 + (ln >> 4) * 8 + (j < 8 ? j : j + 8);
    int n = nt * 16 + (ln & 15);
    w2s[idx] = (_Float16)((k < 40 && n < 20) ? Wf2[k * 20 + n] : 0.0f);
  }
  for (int idx = tid; idx < 512; idx += 256) {
    int ln = idx >> 4, j = idx & 15;
    int k = (ln >> 4) * 8 + (j < 8 ? j : j + 8);
    int n = ln & 15;
    w3s[idx] = (_Float16)((k < 20 && n < 4) ? Wf3[k * 4 + n] : 0.0f);
  }
  if (tid < 48) bias1[tid] = (tid < 40) ? bf1[tid] : 0.0f;
  if (tid < 32) bias2[tid] = (tid < 20) ? bf2[tid] : 0.0f;
  __syncthreads();

  const int wave = tid >> 5;
  const int lane = tid & 31;
  const int hh   = lane >> 4;
  const int n16  = lane & 15;
  _Float16* zw = &ztile[wave][0];
  const int r0 = blockIdx.x * 128 + wave * 16;

  // ---- gather 16 rows x 100 features (pad to 128) as f16, packed stores ----
  {
    int r = lane >> 1, part = lane & 1;      // 2 lanes per row
    int gr = r0 + r;
    bool ok = gr < P;
    int nid = ok ? (part ? node2[gr] : node1[gr]) : 0;
    const float* hp = allh + (long)nid * 50; // 8B-aligned (200B row stride)
    unsigned* zrow = (unsigned*)&zw[r * 128];
    int dw0 = part * 25;                     // dword offset: cols part*50 ..
#pragma unroll
    for (int kk = 0; kk < 25; ++kk) {
      float2 w = ok ? ((const float2*)hp)[kk] : make_float2(0.0f, 0.0f);
      union { _Float16 h[2]; unsigned u; } pk;
      pk.h[0] = (_Float16)w.x; pk.h[1] = (_Float16)w.y;
      zrow[dw0 + kk] = pk.u;
    }
    if (part) {                              // zero pad cols 100..127
#pragma unroll
      for (int kk = 0; kk < 14; ++kk) zrow[50 + kk] = 0u;
    }
  }
  __syncthreads();

  // ---- layer 1: [16x128]@[128x48] ----
  v8f acc1[3];
#pragma unroll
  for (int nt = 0; nt < 3; ++nt) {
    v8f c = {0.f, 0.f, 0.f, 0.f, 0.f, 0.f, 0.f, 0.f};
#pragma unroll
    for (int kt = 0; kt < 4; ++kt) {
      v16h a, b;
      const _Float16* zr = &zw[n16 * 128 + kt * 32 + hh * 8];
#pragma unroll
      for (int j = 0; j < 8; ++j) { a[j] = zr[j]; a[8 + j] = zr[16 + j]; }
      const _Float16* wr = &w1s[((kt * 3 + nt) * 32 + lane) * 16];
#pragma unroll
      for (int j = 0; j < 16; ++j) b[j] = wr[j];
      c = __builtin_amdgcn_wmma_f32_16x16x32_f16(false, a, false, b, (short)0, c, false, false);
    }
    acc1[nt] = c;
  }
  __syncthreads();

  // ---- stage layer-2 activations (16x64 f16, bias fused, zero padded) ----
  _Float16* z2 = zw;
#pragma unroll
  for (int nt = 0; nt < 3; ++nt) {
    int col = nt * 16 + n16;
    float bc = bias1[col];
#pragma unroll
    for (int v = 0; v < 8; ++v)
      z2[(v + 8 * hh) * 64 + col] = (_Float16)(acc1[nt][v] + bc);
  }
#pragma unroll
  for (int v = 0; v < 8; ++v)
    z2[(v + 8 * hh) * 64 + 48 + n16] = (_Float16)0.0f;
  __syncthreads();

  // ---- layer 2: [16x64]@[64x32] ----
  v8f acc2[2];
#pragma unroll
  for (int nt = 0; nt < 2; ++nt) {
    v8f c = {0.f, 0.f, 0.f, 0.f, 0.f, 0.f, 0.f, 0.f};
#pragma unroll
    for (int kt = 0; kt < 2; ++kt) {
      v16h a, b;
      const _Float16* zr = &z2[n16 * 64 + kt * 32 + hh * 8];
#pragma unroll
      for (int j = 0; j < 8; ++j) { a[j] = zr[j]; a[8 + j] = zr[16 + j]; }
      const _Float16* wr = &w2s[((kt * 2 + nt) * 32 + lane) * 16];
#pragma unroll
      for (int j = 0; j < 16; ++j) b[j] = wr[j];
      c = __builtin_amdgcn_wmma_f32_16x16x32_f16(false, a, false, b, (short)0, c, false, false);
    }
    acc2[nt] = c;
  }
  __syncthreads();

  // ---- stage layer-3 activations (16x32 f16, bias fused) ----
  _Float16* zl3 = zw + 16 * 64;
#pragma unroll
  for (int nt = 0; nt < 2; ++nt) {
    int col = nt * 16 + n16;
    float bc = bias2[col];
#pragma unroll
    for (int v = 0; v < 8; ++v)
      zl3[(v + 8 * hh) * 32 + col] = (_Float16)(acc2[nt][v] + bc);
  }
  __syncthreads();

  // ---- layer 3: [16x32]@[32x16] (4 real cols) ----
  v8f c3 = {0.f, 0.f, 0.f, 0.f, 0.f, 0.f, 0.f, 0.f};
  {
    v16h a, b;
    const _Float16* zr = &zl3[n16 * 32 + hh * 8];
#pragma unroll
    for (int j = 0; j < 8; ++j) { a[j] = zr[j]; a[8 + j] = zr[16 + j]; }
    const _Float16* wr = &w3s[lane * 16];
#pragma unroll
    for (int j = 0; j < 16; ++j) b[j] = wr[j];
    c3 = __builtin_amdgcn_wmma_f32_16x16x32_f16(false, a, false, b, (short)0, c3, false, false);
  }

  if (n16 < 4) {
    float bc = bf3[n16];
#pragma unroll
    for (int v = 0; v < 8; ++v) {
      int gr = r0 + v + 8 * hh;
      if (gr < P) zout[gr * 4 + n16] = c3[v] + bc;
    }
  }
}

// ---------------- column softmax over axis 0 ----------------

__global__ void colmax_part_kernel(const float* __restrict__ z, float* __restrict__ pmax, int P) {
  __shared__ float sm[256 * 4];
  float m0 = -3.0e38f, m1 = -3.0e38f, m2 = -3.0e38f, m3 = -3.0e38f;
  for (long i = (long)blockIdx.x * blockDim.x + threadIdx.x; i < P;
       i += (long)gridDim.x * blockDim.x) {
    const float* r = z + i * 4;
    m0 = fmaxf(m0, r[0]); m1 = fmaxf(m1, r[1]);
    m2 = fmaxf(m2, r[2]); m3 = fmaxf(m3, r[3]);
  }
  sm[threadIdx.x * 4 + 0] = m0; sm[threadIdx.x * 4 + 1] = m1;
  sm[threadIdx.x * 4 + 2] = m2; sm[threadIdx.x * 4 + 3] = m3;
  __syncthreads();
  for (int s = 128; s > 0; s >>= 1) {
    if ((int)threadIdx.x < s)
      for (int c = 0; c < 4; ++c)
        sm[threadIdx.x * 4 + c] = fmaxf(sm[threadIdx.x * 4 + c], sm[(threadIdx.x + s) * 4 + c]);
    __syncthreads();
  }
  if (threadIdx.x == 0)
    for (int c = 0; c < 4; ++c) pmax[blockIdx.x * 4 + c] = sm[c];
}

__global__ void colsum_part_kernel(const float* __restrict__ z, const float* __restrict__ cmx,
                                   float* __restrict__ psum, int P) {
  __shared__ float sm[256 * 4];
  float c0 = cmx[0], c1 = cmx[1], c2 = cmx[2], c3 = cmx[3];
  float s0 = 0.f, s1 = 0.f, s2 = 0.f, s3 = 0.f;
  for (long i = (long)blockIdx.x * blockDim.x + threadIdx.x; i < P;
       i += (long)gridDim.x * blockDim.x) {
    const float* r = z + i * 4;
    s0 += expf(r[0] - c0); s1 += expf(r[1] - c1);
    s2 += expf(r[2] - c2); s3 += expf(r[3] - c3);
  }
  sm[threadIdx.x * 4 + 0] = s0; sm[threadIdx.x * 4 + 1] = s1;
  sm[threadIdx.x * 4 + 2] = s2; sm[threadIdx.x * 4 + 3] = s3;
  __syncthreads();
  for (int s = 128; s > 0; s >>= 1) {
    if ((int)threadIdx.x < s)
      for (int c = 0; c < 4; ++c)
        sm[threadIdx.x * 4 + c] += sm[(threadIdx.x + s) * 4 + c];
    __syncthreads();
  }
  if (threadIdx.x == 0)
    for (int c = 0; c < 4; ++c) psum[blockIdx.x * 4 + c] = sm[c];
}

__global__ void colreduce_fin_kernel(const float* __restrict__ part, float* __restrict__ out4,
                                     int nb, int isMax) {
  int c = threadIdx.x;
  if (c < 4) {
    if (isMax) {
      float m = -3.0e38f;
      for (int b = 0; b < nb; ++b) m = fmaxf(m, part[b * 4 + c]);
      out4[c] = m;
    } else {
      float s = 0.f;
      for (int b = 0; b < nb; ++b) s += part[b * 4 + c];
      out4[c] = s;
    }
  }
}

__global__ void softmax_out_kernel(const float* __restrict__ z, const float* __restrict__ cmx,
                                   const float* __restrict__ csm, float* __restrict__ out, int P) {
  long t = (long)blockIdx.x * blockDim.x + threadIdx.x;
  if (t < (long)P * 4) {
    int c = (int)(t & 3);
    out[t] = expf(z[t] - cmx[c]) / csm[c];
  }
}

// ---------------- launcher ----------------

extern "C" void kernel_launch(void* const* d_in, const int* in_sizes, int n_in,
                              void* d_out, int out_size, void* d_ws, size_t ws_size,
                              hipStream_t stream) {
  (void)in_sizes; (void)n_in; (void)out_size; (void)ws_size;
  const float* x    = (const float*)d_in[0];
  const float* bank = (const float*)d_in[1];
  const float* W1   = (const float*)d_in[2];
  const float* b1   = (const float*)d_in[3];
  const float* W2   = (const float*)d_in[4];
  const float* b2   = (const float*)d_in[5];
  const float* Wb   = (const float*)d_in[6];
  const float* bb   = (const float*)d_in[7];
  const float* Wf1  = (const float*)d_in[8];
  const float* bf1  = (const float*)d_in[9];
  const float* Wf2  = (const float*)d_in[10];
  const float* bf2  = (const float*)d_in[11];
  const float* Wf3  = (const float*)d_in[12];
  const float* bf3  = (const float*)d_in[13];
  const int* ei     = (const int*)d_in[14];
  const int* node1  = (const int*)d_in[15];
  const int* node2  = (const int*)d_in[16];
  float* out = (float*)d_out;

  const int N = 100000, E = 1600000, NB = 1000, P = 500000;
  const int* srcv = ei;
  const int* dstv = ei + E;

  char* ws = (char*)d_ws;
  float* dinv = (float*)ws;                                          // N floats
  float* bufB = (float*)(ws + (size_t)(4u << 20));                   // 24 MB (xw / z3)
  float* bufC = (float*)(ws + (size_t)(4u << 20) + 24000000u);       // 24 MB (acc / allh)
  float* pmax = (float*)(ws + (size_t)(4u << 20) + 48000000u);
  float* psum = pmax + 1024;
  float* cmx  = psum + 1024;
  float* csm  = cmx + 4;

  auto cdiv = [](long a, long b) { return (int)((a + b - 1) / b); };

  // symmetric normalization terms
  fill_kernel<<<cdiv(N, 256), 256, 0, stream>>>(dinv, 1.0f, N);   // self-loop degree
  deg_kernel<<<cdiv(E, 256), 256, 0, stream>>>(dstv, dinv, E);
  rsqrt_kernel<<<cdiv(N, 256), 256, 0, stream>>>(dinv, N);

  // GCN layer 1: xw1 = x@W1 ; acc = scatter ; h1 = acc + self + b1 (in bufC)
  dense_kernel<<<cdiv((long)N * 60, 256), 256, 27 * 60 * 4, stream>>>(x, W1, nullptr, bufB, N, 27, 60, 0);
  fill_kernel<<<cdiv((long)N * 60, 256), 256, 0, stream>>>(bufC, 0.0f, (long)N * 60);
  scatter_kernel<<<cdiv((long)E * 32, 256), 256, 0, stream>>>(srcv, dstv, dinv, bufB, bufC, 60, E);
  finish_kernel<<<cdiv((long)N * 60, 256), 256, 0, stream>>>(bufC, bufB, dinv, b1, N, 60);

  // GCN layer 2: xw2 = h1@W2 (bufB) ; acc in bufC ; h2 in bufC[0:N*50]
  dense_kernel<<<cdiv((long)N * 50, 256), 256, 60 * 50 * 4, stream>>>(bufC, W2, nullptr, bufB, N, 60, 50, 0);
  fill_kernel<<<cdiv((long)N * 50, 256), 256, 0, stream>>>(bufC, 0.0f, (long)N * 50);
  scatter_kernel<<<cdiv((long)E * 32, 256), 256, 0, stream>>>(srcv, dstv, dinv, bufB, bufC, 50, E);
  finish_kernel<<<cdiv((long)N * 50, 256), 256, 0, stream>>>(bufC, bufB, dinv, b2, N, 50);

  // bank rows: allh[N:N+NB] = bank@Wb + bb
  dense_kernel<<<cdiv((long)NB * 50, 256), 256, 27 * 50 * 4, stream>>>(bank, Wb, bb, bufC + (long)N * 50, NB, 27, 50, 1);

  // fused gather + WMMA MLP -> z3[P,4] in bufB
  float* z3 = bufB;
  mlp_wmma_kernel<<<cdiv(P, 128), 256, 0, stream>>>(bufC, node1, node2,
                                                    Wf1, bf1, Wf2, bf2, Wf3, bf3, z3, P);

  // softmax over axis 0
  colmax_part_kernel<<<256, 256, 0, stream>>>(z3, pmax, P);
  colreduce_fin_kernel<<<1, 32, 0, stream>>>(pmax, cmx, 256, 1);
  colsum_part_kernel<<<256, 256, 0, stream>>>(z3, cmx, psum, P);
  colreduce_fin_kernel<<<1, 32, 0, stream>>>(psum, csm, 256, 0);
  softmax_out_kernel<<<cdiv((long)P * 4, 256), 256, 0, stream>>>(z3, cmx, csm, out, P);
}
